// DecisionTreeModel_12051678232739
// MI455X (gfx1250) — matrix operand order repaired and checked by reading, not compiled, not run on hardware
//
#include <hip/hip_runtime.h>
#include <hip/hip_fp16.h>

// ---------------------------------------------------------------------------
// Soft decision tree, fused for MI455X (gfx1250, wave32, WMMA f16->f32).
//   d      = sigmoid(X @ thr^T)           (GEMM1, 68.7 GFLOP)
//   probs  = tree path products           (cheap, LDS gathers)
//   logits = probs @ leaf                 (GEMM2, 68.7 GFLOP)
//   out    = softmax(logits)              (f32 throughout)
// Only X (128MB) in and out (128MB) written to HBM; everything else in LDS.
// Workspace: needs 4 MB (thr as f16 padded, leaf^T as f16), L2-resident.
// B fragments are register double-buffered to hide L2 latency.
// ---------------------------------------------------------------------------

typedef __attribute__((ext_vector_type(16))) _Float16 v16h;
typedef __attribute__((ext_vector_type(4)))  _Float16 v4h;
typedef __attribute__((ext_vector_type(8)))  float    v8f;

#define DIM        1024
#define N_INTERNAL 1023
#define M_TILE     64
#define BLOCK      512          // 16 waves
#define NWAVE      16
#define LSTRIDE    1032         // f16 elements per LDS row (pad 8 -> 4-bank skew)

// LDS byte offsets
#define XBUF_OFF   0                       // 64 x 1032 f16 : x tile, later probs
#define DBUF_OFF   (M_TILE * LSTRIDE * 2)  // 64 x 1032 f16 : sigmoid outputs
#define AUX_OFF    (2 * M_TILE * LSTRIDE * 2)
#define SMEM_BYTES (AUX_OFF + 8192)        // aux: temp5 (64x32 f32) / red+rowv

union Frag { v16h v; uint4 q[2]; };

static __device__ __forceinline__ v8f wmma_f16(const Frag& a, const Frag& b, v8f c) {
    return __builtin_amdgcn_wmma_f32_16x16x32_f16(false, a.v, false, b.v,
                                                  (short)0, c, false, false);
}

// One K=1024 pass over 4 M-tiles for a single 16-col N-tile.
// B fragments double-buffered in registers: B(ks+1) in flight while B(ks)
// feeds 4 WMMAs -> the s_wait before the first WMMA is loadcnt<=2, not 0.
static __device__ __forceinline__ void gemm_ktile(const uint4* __restrict__ bq,
                                                  const _Float16* __restrict__ aBase,
                                                  v8f (&acc)[4]) {
    Frag b;
    b.q[0] = bq[0];
    b.q[1] = bq[2];
    #pragma unroll 4
    for (int ks = 0; ks < 32; ++ks) {
        Frag nb;
        if (ks + 1 < 32) {                       // next B: 2 global_load_b128
            nb.q[0] = bq[(ks + 1) * 4];
            nb.q[1] = bq[(ks + 1) * 4 + 2];
        }
        Frag a0, a1, a2, a3;                     // 8 ds_load_b128, then 4 wmma
        const _Float16* ap = aBase + ks * 32;
        a0.q[0] = *(const uint4*)(ap);
        a0.q[1] = *(const uint4*)(ap + 16);
        a1.q[0] = *(const uint4*)(ap + 16 * LSTRIDE);
        a1.q[1] = *(const uint4*)(ap + 16 * LSTRIDE + 16);
        a2.q[0] = *(const uint4*)(ap + 32 * LSTRIDE);
        a2.q[1] = *(const uint4*)(ap + 32 * LSTRIDE + 16);
        a3.q[0] = *(const uint4*)(ap + 48 * LSTRIDE);
        a3.q[1] = *(const uint4*)(ap + 48 * LSTRIDE + 16);
        acc[0] = wmma_f16(a0, b, acc[0]);
        acc[1] = wmma_f16(a1, b, acc[1]);
        acc[2] = wmma_f16(a2, b, acc[2]);
        acc[3] = wmma_f16(a3, b, acc[3]);
        if (ks + 1 < 32) b = nb;
    }
}

// --- prep: thresholds f32[1023][1024] -> f16[1024][1024] (row 1023 zeroed) ---
__global__ void k_convert_thr(const float* __restrict__ thr,
                              _Float16* __restrict__ thrH) {
    int idx = blockIdx.x * blockDim.x + threadIdx.x;   // over 1024*1024
    int n = idx >> 10;
    float v = (n < N_INTERNAL) ? thr[idx] : 0.0f;
    thrH[idx] = (_Float16)v;
}

// --- prep: leaf f32[1024 leaves][1024 out] -> leafT f16[out][leaf] ----------
__global__ void k_transpose_leaf(const float* __restrict__ leaf,
                                 _Float16* __restrict__ leafT) {
    __shared__ float tile[32][33];
    int bx = blockIdx.x, by = blockIdx.y;
    #pragma unroll
    for (int i = 0; i < 4; ++i) {
        int l = by * 32 + threadIdx.y + i * 8;
        int o = bx * 32 + threadIdx.x;
        tile[threadIdx.y + i * 8][threadIdx.x] = leaf[(size_t)l * DIM + o];
    }
    __syncthreads();
    #pragma unroll
    for (int i = 0; i < 4; ++i) {
        int o = bx * 32 + threadIdx.y + i * 8;
        int l = by * 32 + threadIdx.x;
        leafT[(size_t)o * DIM + l] = (_Float16)tile[threadIdx.x][threadIdx.y + i * 8];
    }
}

// ---------------------------- fused main kernel -----------------------------
__global__ __launch_bounds__(BLOCK)
void k_fused_tree(const float* __restrict__ x,
                  const _Float16* __restrict__ thrH,
                  const _Float16* __restrict__ leafT,
                  float* __restrict__ out) {
    extern __shared__ char smem[];
    _Float16* xbuf  = (_Float16*)(smem + XBUF_OFF);   // x tile -> probs
    _Float16* dbuf  = (_Float16*)(smem + DBUF_OFF);   // sigmoid d
    float*    temp5 = (float*)(smem + AUX_OFF);       // [64][32] level-5 prefix
    float*    red   = (float*)(smem + AUX_OFF);       // [64][16] partials (reuse)
    float*    rowv  = (float*)(smem + AUX_OFF + 64 * NWAVE * 4); // [64]

    const int tid  = threadIdx.x;
    const int wave = tid >> 5;
    const int lane = tid & 31;
    const int l16  = lane & 15;
    const int half = lane >> 4;           // 0: lanes 0-15, 1: lanes 16-31
    const int row0 = blockIdx.x * M_TILE;
    const int nwBase = wave * 64;         // 64 output columns per wave

    // Per-lane LDS base for A fragments (ISA A layout: lane=M row,
    // K chunks {k0..k0+7, k0+16..k0+23} for lanes 0-15, +8 for lanes 16-31).
    const _Float16* aBase = xbuf + l16 * LSTRIDE + half * 8;

    // ---- Phase 0: load X tile f32 -> f16 LDS (coalesced float4) ----
    {
        const float4* xv = (const float4*)(x + (size_t)row0 * DIM);
        #pragma unroll
        for (int i = 0; i < 32; ++i) {
            int idx = tid + i * BLOCK;            // 0 .. 16383
            int m  = idx >> 8;
            int c4 = idx & 255;
            float4 v = xv[(size_t)m * 256 + c4];
            v4h h = { (_Float16)v.x, (_Float16)v.y, (_Float16)v.z, (_Float16)v.w };
            *(v4h*)(xbuf + m * LSTRIDE + c4 * 4) = h;
        }
    }
    __syncthreads();

    // ---- Phase 1: GEMM1  d = sigmoid(X @ thr^T), K=1024 ----
    // B frag (ISA layout): lane<16 -> N=n0+lane, K={k0..k0+7, k0+16..k0+23} ; lane>=16 -> +8.
    #pragma unroll 1
    for (int nt = 0; nt < 4; ++nt) {
        int n0 = nwBase + nt * 16;
        v8f acc[4] = {};
        const uint4* bq = (const uint4*)(thrH + (size_t)(n0 + l16) * DIM + half * 8);
        gemm_ktile(bq, aBase, acc);
        #pragma unroll
        for (int mt = 0; mt < 4; ++mt) {
            #pragma unroll
            for (int r = 0; r < 8; ++r) {
                int m = mt * 16 + r + half * 8;        // C layout: VGPR r -> M=r / r+8
                float d = 1.0f / (1.0f + __expf(-acc[mt][r]));
                dbuf[m * LSTRIDE + n0 + l16] = (_Float16)d;
            }
        }
    }
    __syncthreads();

    // ---- Phase 2a: 5-level prefix products temp5[m][j], j = top-5 bits ----
    #pragma unroll
    for (int i = 0; i < 4; ++i) {
        int idx = tid + i * BLOCK;                    // 0 .. 2047
        int m = idx >> 5, j = idx & 31;
        float p = 1.0f;
        #pragma unroll
        for (int k = 0; k < 5; ++k) {
            int node = ((1 << k) - 1) + (j >> (5 - k));
            float d = (float)dbuf[m * LSTRIDE + node];
            p *= ((j >> (4 - k)) & 1) ? (1.0f - d) : d;
        }
        temp5[m * 32 + j] = p;
    }
    __syncthreads();

    // ---- Phase 2b: leaf probs -> overwrite xbuf (x no longer needed) ----
    #pragma unroll 2
    for (int i = 0; i < 128; ++i) {
        int idx = tid + i * BLOCK;                    // 0 .. 65535
        int m = idx >> 10, l = idx & 1023;
        float p = temp5[m * 32 + (l >> 5)];
        #pragma unroll
        for (int k = 5; k < 10; ++k) {
            int node = ((1 << k) - 1) + (l >> (10 - k));
            float d = (float)dbuf[m * LSTRIDE + node];
            p *= ((l >> (9 - k)) & 1) ? (1.0f - d) : d;
        }
        xbuf[m * LSTRIDE + l] = (_Float16)p;
    }
    __syncthreads();

    // ---- Phase 3: GEMM2  logits = probs @ leaf (all 16 acc sets in regs) ----
    v8f lacc[4][4];
    #pragma unroll
    for (int nt = 0; nt < 4; ++nt)
        #pragma unroll
        for (int mt = 0; mt < 4; ++mt) lacc[nt][mt] = (v8f){};
    #pragma unroll
    for (int nt = 0; nt < 4; ++nt) {
        int n0 = nwBase + nt * 16;
        const uint4* bq = (const uint4*)(leafT + (size_t)(n0 + l16) * DIM + half * 8);
        gemm_ktile(bq, aBase, lacc[nt]);
    }
    __syncthreads();   // temp5 region is now free for reduction scratch

    // ---- Phase 4: softmax (f32, register logits + shfl + LDS partials) ----
    #pragma unroll
    for (int mt = 0; mt < 4; ++mt) {
        #pragma unroll
        for (int r = 0; r < 8; ++r) {
            float v = lacc[0][mt][r];
            #pragma unroll
            for (int nt = 1; nt < 4; ++nt) v = fmaxf(v, lacc[nt][mt][r]);
            #pragma unroll
            for (int s = 1; s < 16; s <<= 1) v = fmaxf(v, __shfl_xor(v, s, 32));
            if (l16 == 0) red[(mt * 16 + r + half * 8) * NWAVE + wave] = v;
        }
    }
    __syncthreads();
    if (tid < M_TILE) {
        float mx = red[tid * NWAVE];
        for (int w = 1; w < NWAVE; ++w) mx = fmaxf(mx, red[tid * NWAVE + w]);
        rowv[tid] = mx;
    }
    __syncthreads();
    #pragma unroll
    for (int mt = 0; mt < 4; ++mt) {
        #pragma unroll
        for (int r = 0; r < 8; ++r) {
            int m = mt * 16 + r + half * 8;
            float mx = rowv[m];
            float s = 0.0f;
            #pragma unroll
            for (int nt = 0; nt < 4; ++nt) {
                float e = __expf(lacc[nt][mt][r] - mx);
                lacc[nt][mt][r] = e;
                s += e;
            }
            #pragma unroll
            for (int sh = 1; sh < 16; sh <<= 1) s += __shfl_xor(s, sh, 32);
            if (l16 == 0) red[m * NWAVE + wave] = s;
        }
    }
    __syncthreads();
    if (tid < M_TILE) {
        float s = 0.0f;
        for (int w = 0; w < NWAVE; ++w) s += red[tid * NWAVE + w];
        rowv[tid] = 1.0f / s;
    }
    __syncthreads();

    // ---- Phase 5: scale + store f32 output ----
    #pragma unroll
    for (int nt = 0; nt < 4; ++nt) {
        int n0 = nwBase + nt * 16;
        #pragma unroll
        for (int mt = 0; mt < 4; ++mt) {
            #pragma unroll
            for (int r = 0; r < 8; ++r) {
                int m = mt * 16 + r + half * 8;
                out[(size_t)(row0 + m) * DIM + n0 + l16] = lacc[nt][mt][r] * rowv[m];
            }
        }
    }
}

extern "C" void kernel_launch(void* const* d_in, const int* in_sizes, int n_in,
                              void* d_out, int out_size, void* d_ws, size_t ws_size,
                              hipStream_t stream) {
    const float* x    = (const float*)d_in[0];
    const float* thr  = (const float*)d_in[1];   // [1023][1024]
    const float* leaf = (const float*)d_in[2];   // [1024][1024]
    float* out = (float*)d_out;

    // workspace: thrH f16 [1024][1024] (2MB) | leafT f16 [1024][1024] (2MB)
    _Float16* thrH  = (_Float16*)d_ws;
    _Float16* leafT = (_Float16*)((char*)d_ws + (size_t)2 * 1024 * 1024);

    k_convert_thr<<<(DIM * DIM) / 256, 256, 0, stream>>>(thr, thrH);
    k_transpose_leaf<<<dim3(32, 32), dim3(32, 8), 0, stream>>>(leaf, leafT);
    k_fused_tree<<<32768 / M_TILE, BLOCK, SMEM_BYTES, stream>>>(x, thrH, leafT, out);
}
